// SingleHeadSelfAttention_68667937128837
// MI455X (gfx1250) — matrix operand order repaired and checked by reading, not compiled, not run on hardware
//
#include <hip/hip_runtime.h>
#include <math.h>

// Problem constants (reference: B=4, S=4096, D=256, fp32)
#define BB 4
#define SS 4096
#define DD 256

typedef __attribute__((ext_vector_type(16))) __bf16 v16bf;
typedef __attribute__((ext_vector_type(8)))  float  v8f;
typedef int v4i __attribute__((ext_vector_type(4)));

union V16 {
  v16bf v;
  uint4 q[2];            // q[0] = elements 0..7, q[1] = elements 8..15
  unsigned short s[16];
};

// fp32 -> bf16 round-to-nearest-even
__device__ __forceinline__ unsigned short f2bf(float f) {
  unsigned u = __builtin_bit_cast(unsigned, f);
  u += 0x7FFFu + ((u >> 16) & 1u);
  return (unsigned short)(u >> 16);
}

__device__ __forceinline__ v8f wmma_bf16(v16bf a, v16bf b, v8f c) {
  return __builtin_amdgcn_wmma_f32_16x16x32_bf16(
      /*neg_a=*/false, a, /*neg_b=*/false, b,
      /*c_mod=*/(short)0, c, /*reuse_a=*/false, /*reuse_b=*/false);
}

// ---------------------------------------------------------------------------
// Async global->LDS copy (CDNA5 GLOBAL_LOAD_ASYNC_TO_LDS_B128, ASYNCcnt).
// Probe-discovered signature: (v4i as(1)* src, v4i as(3)* dst, Imm off, Imm cpol)
// Falls back to sync load + ds_store if the builtin is unavailable.
// ---------------------------------------------------------------------------
#if __has_builtin(__builtin_amdgcn_global_load_async_to_lds_b128)
#define HAVE_ASYNC 1
__device__ __forceinline__ void async_cp16(const unsigned short* g, unsigned short* l) {
  __builtin_amdgcn_global_load_async_to_lds_b128(
      (__attribute__((address_space(1))) v4i*)g,
      (__attribute__((address_space(3))) v4i*)l, 0, 0);
}
#else
#define HAVE_ASYNC 0
__device__ __forceinline__ void async_cp16(const unsigned short* g, unsigned short* l) {
  *(uint4*)l = *(const uint4*)g;   // global_load_b128 + ds_store_b128
}
#endif

#if HAVE_ASYNC
#if __has_builtin(__builtin_amdgcn_s_wait_asynccnt)
#define WAIT_ASYNC(n) __builtin_amdgcn_s_wait_asynccnt(n)
#else
#define WAIT_ASYNC(n) asm volatile("s_wait_asynccnt " #n ::: "memory")
#endif
#else
#define WAIT_ASYNC(n) do {} while (0)
#endif

// LDS tile geometry (in ushorts), padded against bank conflicts.
#define KROW 264                   // 256 K-d elements (512B) padded to 528B
#define VROW 40                    // 32 keys (64B) padded to 80B
#define KTILE (32 * KROW)          // 8448  : K block, 32 keys x 256 d
#define VTILE (256 * VROW)         // 10240 : V block, 256 feats x 32 keys
#define BUFSZ (KTILE + VTILE)      // 18688 per buffer
#define PBASE (2 * BUFSZ)          // 37376
#define LDS_TOT (PBASE + 4 * 512)  // + per-wave P staging = 39424 us = 78.8KB

// ---------------------------------------------------------------------------
// Kernel 1: fp32 -> bf16 conversion
// ---------------------------------------------------------------------------
__global__ void cvt_kernel(const float* __restrict__ in,
                           unsigned short* __restrict__ out, int n) {
  int i = blockIdx.x * blockDim.x + threadIdx.x;
  if (i < n) out[i] = f2bf(in[i]);
}

// ---------------------------------------------------------------------------
// Kernel 2: QKV projection (one wave = 16x64 output strip of one matrix).
// Q scaled by 1/sqrt(D); V stored transposed as Vt[b, feat, s].
// ---------------------------------------------------------------------------
__global__ __launch_bounds__(128) void qkv_kernel(
    const unsigned short* __restrict__ xb,
    const unsigned short* __restrict__ Wqb,
    const unsigned short* __restrict__ Wkb,
    const unsigned short* __restrict__ Wvb,
    unsigned short* __restrict__ Qb,
    unsigned short* __restrict__ Kb,
    unsigned short* __restrict__ Vtb) {
  const int wid  = blockIdx.x * 4 + (threadIdx.x >> 5);
  const int lane = threadIdx.x & 31;
  const int ln   = lane & 15;
  const int h    = lane >> 4;

  const int cs  = wid & 3;
  const int rt  = (wid >> 2) & 255;
  const int mat = (wid >> 10) % 3;
  const int b   = wid / 3072;

  const unsigned short* W = (mat == 0) ? Wqb : (mat == 1) ? Wkb : Wvb;

  const unsigned short* xrow = xb + ((size_t)(b * SS) + rt * 16 + ln) * DD;
  V16 A[8];
#pragma unroll
  for (int ks = 0; ks < 8; ++ks) {
    A[ks].q[0] = *(const uint4*)(xrow + ks * 32 + 8 * h);
    A[ks].q[1] = *(const uint4*)(xrow + ks * 32 + 16 + 8 * h);
  }

  v8f C[4] = {};
#pragma unroll
  for (int ks = 0; ks < 8; ++ks) {
#pragma unroll
    for (int nt = 0; nt < 4; ++nt) {
      const int n = cs * 64 + nt * 16 + ln;
      const unsigned short* wr = W + (size_t)n * DD + ks * 32 + 8 * h;
      V16 Bv;
      Bv.q[0] = *(const uint4*)(wr);
      Bv.q[1] = *(const uint4*)(wr + 16);
      C[nt] = wmma_bf16(A[ks].v, Bv.v, C[nt]);
    }
  }

#pragma unroll
  for (int nt = 0; nt < 4; ++nt) {
#pragma unroll
    for (int r = 0; r < 8; ++r) {
      const int row  = rt * 16 + r + 8 * h;
      const int feat = cs * 64 + nt * 16 + ln;
      const float v = C[nt][r];
      if (mat == 0)
        Qb[((size_t)(b * SS) + row) * DD + feat] = f2bf(v * 0.0625f);
      else if (mat == 1)
        Kb[((size_t)(b * SS) + row) * DD + feat] = f2bf(v);
      else
        Vtb[((size_t)(b * DD) + feat) * SS + row] = f2bf(v);
    }
  }
}

// ---------------------------------------------------------------------------
// Async-stage one 32-key K/V block into LDS buffer `buf`.
// 128 threads x 16 chunks x 16B = 32KB (16KB K + 16KB V), padded rows.
// ---------------------------------------------------------------------------
__device__ __forceinline__ void stage_kv(unsigned short* lds, int buf, int k0,
                                         int b, int tid,
                                         const unsigned short* __restrict__ Kb,
                                         const unsigned short* __restrict__ Vtb) {
  unsigned short* kbase = lds + buf * BUFSZ;
  unsigned short* vbase = kbase + KTILE;
#pragma unroll
  for (int i = 0; i < 8; ++i) {            // K: 32 rows x 512B, 1024 chunks
    const int c = tid + i * 128;
    const int row = c >> 5, col = c & 31;  // col in 16B units
    async_cp16(Kb + ((size_t)(b * SS) + k0 + row) * DD + col * 8,
               kbase + row * KROW + col * 8);
  }
#pragma unroll
  for (int i = 0; i < 8; ++i) {            // V: 256 rows x 64B, 1024 chunks
    const int c = tid + i * 128;
    const int row = c >> 2, sub = c & 3;
    async_cp16(Vtb + ((size_t)(b * DD) + row) * SS + k0 + sub * 8,
               vbase + row * VROW + sub * 8);
  }
}

// ---------------------------------------------------------------------------
// Kernel 3: causal flash attention, cooperative workgroup version.
// Block = 4 waves = 64 queries of one batch; K/V 32-key blocks staged into
// LDS with async copies, double buffered, shared by all 4 waves.
// ---------------------------------------------------------------------------
__global__ __launch_bounds__(128) void attn_kernel(
    const unsigned short* __restrict__ Qb,
    const unsigned short* __restrict__ Kb,
    const unsigned short* __restrict__ Vtb,
    float* __restrict__ out) {
  __shared__ unsigned short lds[LDS_TOT];

  const int tid  = threadIdx.x;
  const int wave = tid >> 5;
  const int lane = tid & 31;
  const int ln   = lane & 15;
  const int h    = lane >> 4;

  const int qblk  = blockIdx.x & 63;   // 64-query block index
  const int b     = blockIdx.x >> 6;
  const int qbase = qblk * 64 + wave * 16;

  const float NEG_INF = -__builtin_inff();

  // Preload Q tile (pre-scaled by 1/sqrt(D)) in A layout
  const unsigned short* qrow = Qb + ((size_t)(b * SS) + qbase + ln) * DD;
  V16 Aq[8];
#pragma unroll
  for (int ks = 0; ks < 8; ++ks) {
    Aq[ks].q[0] = *(const uint4*)(qrow + ks * 32 + 8 * h);
    Aq[ks].q[1] = *(const uint4*)(qrow + ks * 32 + 16 + 8 * h);
  }

  v8f O[16] = {};
  float m_r[8], l_r[8];
#pragma unroll
  for (int r = 0; r < 8; ++r) { m_r[r] = NEG_INF; l_r[r] = 0.0f; }

  // Key blocks cover [0, qblk*64 + 64); all waves run the same count,
  // the causal mask zeroes out-of-range contributions for lower waves.
  const int nkb = qblk * 2 + 2;

  stage_kv(lds, 0, 0, b, tid, Kb, Vtb);

  for (int kb = 0; kb < nkb; ++kb) {
    const int k0 = kb * 32;

    if (kb + 1 < nkb) {
      stage_kv(lds, (kb + 1) & 1, k0 + 32, b, tid, Kb, Vtb);
      WAIT_ASYNC(16);   // 16 new in flight; everything older (block kb) done
    } else {
      WAIT_ASYNC(0);
    }
    __syncthreads();    // block kb resident for all waves

    const unsigned short* kbuf = lds + (kb & 1) * BUFSZ;
    const unsigned short* vbuf = kbuf + KTILE;

    // ---- scores: two 16x16 C tiles over D=256 (B from LDS) ------------
    v8f Sc[2] = {};
#pragma unroll
    for (int ks = 0; ks < 8; ++ks) {
#pragma unroll
      for (int t = 0; t < 2; ++t) {
        const unsigned short* kr = kbuf + (t * 16 + ln) * KROW + ks * 32 + 8 * h;
        V16 Bv;
        Bv.q[0] = *(const uint4*)kr;
        Bv.q[1] = *(const uint4*)(kr + 16);
        Sc[t] = wmma_bf16(Aq[ks].v, Bv.v, Sc[t]);
      }
    }

    // ---- causal mask + online softmax ---------------------------------
    float alpha[8];
#pragma unroll
    for (int r = 0; r < 8; ++r) {
      const int q = qbase + r + 8 * h;
      if (k0 + ln > q)      Sc[0][r] = NEG_INF;
      if (k0 + 16 + ln > q) Sc[1][r] = NEG_INF;

      float v = fmaxf(Sc[0][r], Sc[1][r]);
#pragma unroll
      for (int off = 1; off < 16; off <<= 1) v = fmaxf(v, __shfl_xor(v, off, 32));
      const float mnew = fmaxf(m_r[r], v);

      const float p0 = __expf(Sc[0][r] - mnew);
      const float p1 = __expf(Sc[1][r] - mnew);
      Sc[0][r] = p0;
      Sc[1][r] = p1;
      float s = p0 + p1;
#pragma unroll
      for (int off = 1; off < 16; off <<= 1) s += __shfl_xor(s, off, 32);

      const float a = __expf(m_r[r] - mnew);
      l_r[r] = l_r[r] * a + s;
      m_r[r] = mnew;
      alpha[r] = a;
    }

#pragma unroll
    for (int j = 0; j < 16; ++j)
#pragma unroll
      for (int r = 0; r < 8; ++r) O[j][r] *= alpha[r];

    // ---- P: C-layout regs -> per-wave LDS -> A-layout regs ------------
    unsigned short* P = lds + PBASE + wave * 512;
#pragma unroll
    for (int r = 0; r < 8; ++r) {
      P[(r + 8 * h) * 32 + ln]      = f2bf(Sc[0][r]);
      P[(r + 8 * h) * 32 + 16 + ln] = f2bf(Sc[1][r]);
    }
    asm volatile("s_wait_dscnt 0" ::: "memory");  // same-wave LDS in-order
    V16 Ap;
    const unsigned short* pr = P + ln * 32;
    Ap.q[0] = *(const uint4*)(pr + 8 * h);
    Ap.q[1] = *(const uint4*)(pr + 16 + 8 * h);

    // ---- O += P(16x32) @ V(32 x 256) (B from LDS) ---------------------
#pragma unroll
    for (int j = 0; j < 16; ++j) {
      const unsigned short* vr = vbuf + (j * 16 + ln) * VROW + 8 * h;
      V16 Bv;
      Bv.q[0] = *(const uint4*)vr;
      Bv.q[1] = *(const uint4*)(vr + 16);
      O[j] = wmma_bf16(Ap.v, Bv.v, O[j]);
    }

    __syncthreads();    // all waves done reading before buffer reuse
  }

  // ---- epilogue: O / l, fp32 store ------------------------------------
#pragma unroll
  for (int j = 0; j < 16; ++j) {
#pragma unroll
    for (int r = 0; r < 8; ++r) {
      const int row = qbase + r + 8 * h;
      out[((size_t)(b * SS) + row) * DD + j * 16 + ln] = O[j][r] / l_r[r];
    }
  }
}

// ---------------------------------------------------------------------------
extern "C" void kernel_launch(void* const* d_in, const int* in_sizes, int n_in,
                              void* d_out, int out_size, void* d_ws, size_t ws_size,
                              hipStream_t stream) {
  const float* x  = (const float*)d_in[0];
  const float* Wq = (const float*)d_in[1];
  const float* Wk = (const float*)d_in[2];
  const float* Wv = (const float*)d_in[3];
  float* out = (float*)d_out;

  char* ws = (char*)d_ws;
  size_t off = 0;
  auto alloc = [&](size_t bytes) {
    char* p = ws + off;
    off += (bytes + 255) & ~(size_t)255;
    return p;
  };
  unsigned short* xb  = (unsigned short*)alloc((size_t)BB * SS * DD * 2);
  unsigned short* Wqb = (unsigned short*)alloc((size_t)DD * DD * 2);
  unsigned short* Wkb = (unsigned short*)alloc((size_t)DD * DD * 2);
  unsigned short* Wvb = (unsigned short*)alloc((size_t)DD * DD * 2);
  unsigned short* Qb  = (unsigned short*)alloc((size_t)BB * SS * DD * 2);
  unsigned short* Kb  = (unsigned short*)alloc((size_t)BB * SS * DD * 2);
  unsigned short* Vtb = (unsigned short*)alloc((size_t)BB * SS * DD * 2);
  (void)ws_size; (void)in_sizes; (void)n_in; (void)out_size;

  const int nx = BB * SS * DD;
  const int nw = DD * DD;
  cvt_kernel<<<(nx + 255) / 256, 256, 0, stream>>>(x, xb, nx);
  cvt_kernel<<<(nw + 255) / 256, 256, 0, stream>>>(Wq, Wqb, nw);
  cvt_kernel<<<(nw + 255) / 256, 256, 0, stream>>>(Wk, Wkb, nw);
  cvt_kernel<<<(nw + 255) / 256, 256, 0, stream>>>(Wv, Wvb, nw);

  qkv_kernel<<<3072, 128, 0, stream>>>(xb, Wqb, Wkb, Wvb, Qb, Kb, Vtb);

  // 4 batches * 64 query-blocks of 64 = 256 workgroups (4 waves each)
  attn_kernel<<<256, 128, 0, stream>>>(Qb, Kb, Vtb, out);
}
